// LibragradAttention_24060406792851
// MI455X (gfx1250) — matrix operand order repaired and checked by reading, not compiled, not run on hardware
//
#include <hip/hip_runtime.h>

typedef unsigned int   u32;
typedef unsigned short u16;
typedef unsigned long long u64;
typedef __attribute__((ext_vector_type(16))) __bf16 v16bf;
typedef __attribute__((ext_vector_type(8)))  float  v8f;
typedef __attribute__((ext_vector_type(4)))  u32    u32x4;
typedef __attribute__((ext_vector_type(2)))  u32    u32x2;
typedef __attribute__((ext_vector_type(4)))  float  f32x4;

union Frag { v16bf v; u32x4 q[2]; u32 u[8]; };
union Acc  { v8f   v; float f[8]; };

__device__ __forceinline__ u32 packbf2(float a, float b) {
  u32 ua = __builtin_bit_cast(u32, a), ub = __builtin_bit_cast(u32, b);
  ua += 0x7FFFu + ((ua >> 16) & 1u);
  ub += 0x7FFFu + ((ub >> 16) & 1u);
  return (ua >> 16) | (ub & 0xFFFF0000u);
}
__device__ __forceinline__ u16 packbf1(float a) {
  u32 ua = __builtin_bit_cast(u32, a);
  ua += 0x7FFFu + ((ua >> 16) & 1u);
  return (u16)(ua >> 16);
}

// CDNA5 async copy global -> LDS (16 bytes per lane), tracked on ASYNCcnt.
__device__ __forceinline__ u32 lds_addr(const void* p) {
  // low 32 bits of a generic pointer to LDS == byte offset in wave's LDS
  return (u32)(uintptr_t)p;
}
__device__ __forceinline__ void async_ld_b128(u32 lds, const void* g) {
  asm volatile("global_load_async_to_lds_b128 %0, %1, off"
               :: "v"(lds), "v"((u64)(uintptr_t)g) : "memory");
}
__device__ __forceinline__ void wait_async0() {
  asm volatile("s_wait_asynccnt 0x0" ::: "memory");
}

// ---------------------------------------------------------------------------
// GEMM + bias:  C[M,N] = A[M,K] * W[K,N] + bias[N]
// A fp32 or bf16 (template), W/bias fp32, C fp32 or bf16 (template).
// 128x128 tile / workgroup (8 waves), BK=32, ping-pong LDS double buffering.
// bf16-A path stages via global_load_async_to_lds_b128.
// ---------------------------------------------------------------------------
template<bool A_BF16, bool OUT_BF16>
__global__ __launch_bounds__(256) void gemm_bias_kernel(
    const void* __restrict__ Ap, const float* __restrict__ W,
    const float* __restrict__ bias, void* __restrict__ Cp,
    int M, int K, int N)
{
  constexpr int BM = 128, BN = 128, BK = 32, LDA = BK + 8, LDB = BK + 8;
  __shared__ u16 As[2][BM * LDA];   // [row][k]      bf16
  __shared__ u16 Bs[2][BN * LDB];   // [n][k] (W^T)  bf16

  const int tid  = threadIdx.x, lane = tid & 31, wid = tid >> 5;
  const int l16  = lane & 15,   half = lane >> 4;
  const int bm   = blockIdx.y * BM, bn = blockIdx.x * BN;
  const int wm   = (wid & 3) * 32,  wn = (wid >> 2) * 64;
  const int nkt  = K / BK;

  Acc acc[2][4];
  #pragma unroll
  for (int i = 0; i < 2; i++)
    #pragma unroll
    for (int j = 0; j < 4; j++)
      #pragma unroll
      for (int r = 0; r < 8; r++) acc[i][j].f[r] = 0.f;

  // ---- prologue: stage tile 0 into buffer 0 ----
  if constexpr (A_BF16) {
    #pragma unroll
    for (int j = 0; j < 2; j++) {
      int f = tid + j * 256;              // 512 x 16B ops
      int row = f >> 2, c8 = (f & 3) * 8;
      async_ld_b128(lds_addr(&As[0][row * LDA + c8]),
                    (const u16*)Ap + (size_t)(bm + row) * K + c8);
    }
  } else {
    #pragma unroll
    for (int j = 0; j < 4; j++) {
      int f = tid + j * 256;              // 1024 x 4-elem groups
      int row = f >> 3, c4 = (f & 7) * 4;
      const f32x4 x = *(const f32x4*)((const float*)Ap +
                                      (size_t)(bm + row) * K + c4);
      u32x2 pk; pk.x = packbf2(x.x, x.y); pk.y = packbf2(x.z, x.w);
      *(u32x2*)&As[0][row * LDA + c4] = pk;
    }
  }
  #pragma unroll
  for (int j = 0; j < 4; j++) {
    int f = tid + j * 256;
    int kr = f >> 5, c4 = (f & 31) * 4;
    const f32x4 w = *(const f32x4*)(W + (size_t)kr * N + bn + c4);
    Bs[0][(c4 + 0) * LDB + kr] = packbf1(w.x);
    Bs[0][(c4 + 1) * LDB + kr] = packbf1(w.y);
    Bs[0][(c4 + 2) * LDB + kr] = packbf1(w.z);
    Bs[0][(c4 + 3) * LDB + kr] = packbf1(w.w);
  }
  if constexpr (A_BF16) wait_async0();
  __syncthreads();

  for (int kt = 0; kt < nkt; kt++) {
    const int p = kt & 1;
    const bool more = (kt + 1 < nkt);
    const int k0n = (kt + 1) * BK;

    // ---- issue next tile's global traffic first (overlaps with WMMA) ----
    f32x4 aR[4], wR[4];
    if (more) {
      if constexpr (A_BF16) {
        #pragma unroll
        for (int j = 0; j < 2; j++) {
          int f = tid + j * 256;
          int row = f >> 2, c8 = (f & 3) * 8;
          async_ld_b128(lds_addr(&As[1 - p][row * LDA + c8]),
                        (const u16*)Ap + (size_t)(bm + row) * K + k0n + c8);
        }
      } else {
        #pragma unroll
        for (int j = 0; j < 4; j++) {
          int f = tid + j * 256;
          int row = f >> 3, c4 = (f & 7) * 4;
          aR[j] = *(const f32x4*)((const float*)Ap +
                                  (size_t)(bm + row) * K + k0n + c4);
        }
      }
      #pragma unroll
      for (int j = 0; j < 4; j++) {
        int f = tid + j * 256;
        int kr = f >> 5, c4 = (f & 31) * 4;
        wR[j] = *(const f32x4*)(W + (size_t)(k0n + kr) * N + bn + c4);
      }
    }
    if (kt + 2 < nkt) {   // spread-prefetch W tile kt+2 into L2
      __builtin_prefetch((const char*)(W + (size_t)((kt + 2) * BK) * N + bn) +
                         (size_t)tid * 256, 0, 3);
    }

    // ---- fragments from buffer p + 8 WMMAs ----
    Frag a[2], bfr[4];
    #pragma unroll
    for (int mt = 0; mt < 2; mt++) {
      int row = wm + mt * 16 + l16;
      a[mt].q[0] = *(const u32x4*)&As[p][row * LDA + half * 8];
      a[mt].q[1] = *(const u32x4*)&As[p][row * LDA + half * 8 + 16];
    }
    #pragma unroll
    for (int nt = 0; nt < 4; nt++) {
      int col = wn + nt * 16 + l16;
      bfr[nt].q[0] = *(const u32x4*)&Bs[p][col * LDB + half * 8];
      bfr[nt].q[1] = *(const u32x4*)&Bs[p][col * LDB + half * 8 + 16];
    }
    #pragma unroll
    for (int mt = 0; mt < 2; mt++)
      #pragma unroll
      for (int nt = 0; nt < 4; nt++)
        acc[mt][nt].v = __builtin_amdgcn_wmma_f32_16x16x32_bf16(
            false, a[mt].v, false, bfr[nt].v, (short)0, acc[mt][nt].v,
            false, false);

    // ---- commit next tile into buffer 1-p ----
    if (more) {
      if constexpr (!A_BF16) {
        #pragma unroll
        for (int j = 0; j < 4; j++) {
          int f = tid + j * 256;
          int row = f >> 3, c4 = (f & 7) * 4;
          u32x2 pk; pk.x = packbf2(aR[j].x, aR[j].y);
          pk.y = packbf2(aR[j].z, aR[j].w);
          *(u32x2*)&As[1 - p][row * LDA + c4] = pk;
        }
      }
      #pragma unroll
      for (int j = 0; j < 4; j++) {
        int f = tid + j * 256;
        int kr = f >> 5, c4 = (f & 31) * 4;
        Bs[1 - p][(c4 + 0) * LDB + kr] = packbf1(wR[j].x);
        Bs[1 - p][(c4 + 1) * LDB + kr] = packbf1(wR[j].y);
        Bs[1 - p][(c4 + 2) * LDB + kr] = packbf1(wR[j].z);
        Bs[1 - p][(c4 + 3) * LDB + kr] = packbf1(wR[j].w);
      }
      if constexpr (A_BF16) wait_async0();
    }
    __syncthreads();
  }

  // ---- epilogue: bias add + store ----
  #pragma unroll
  for (int nt = 0; nt < 4; nt++) {
    int n = bn + wn + nt * 16 + l16;
    float bv = bias[n];
    #pragma unroll
    for (int mt = 0; mt < 2; mt++) {
      #pragma unroll
      for (int r = 0; r < 8; r++) {
        int m = bm + wm + mt * 16 + r + 8 * half;
        float vout = acc[mt][nt].f[r] + bv;
        if constexpr (OUT_BF16)
          ((u16*)Cp)[(size_t)m * N + n] = packbf1(vout);
        else
          ((float*)Cp)[(size_t)m * N + n] = vout;
      }
    }
  }
}

// ---------------------------------------------------------------------------
// Flash attention: per (b, h, 128-query tile). 8 waves, 16 queries each.
// S^T = K * Q^T; exp(S^T) packs straight into P's A-fragment layout.
// Q and K chunks staged with global_load_async_to_lds_b128; V transposed
// via register staging; K/V chunks ping-pong double buffered.
// ---------------------------------------------------------------------------
__global__ __launch_bounds__(256) void attention_kernel(
    const u16* __restrict__ qp, const u16* __restrict__ kp,
    const u16* __restrict__ vp, u16* __restrict__ op)
{
  constexpr int NQc = 2048, NKc = 2048, C = 1024, HD = 64, QT = 128, KCH = 64;
  constexpr int LQ = HD + 8, LK = HD + 8, LV = KCH + 8;
  constexpr int NCH = NKc / KCH;
  __shared__ u16 Qs[QT * LQ];        // [q][d]
  __shared__ u16 Ks[2][KCH * LK];    // [k][d]
  __shared__ u16 Vt[2][HD * LV];     // [d][k] (V^T)

  const int tid = threadIdx.x, lane = tid & 31, wid = tid >> 5;
  const int l16 = lane & 15, half = lane >> 4;
  const int b = blockIdx.y >> 4, h = blockIdx.y & 15;
  const int q0 = blockIdx.x * QT;
  const size_t baseQ  = ((size_t)b * NQc + q0) * C + (size_t)h * HD;
  const size_t baseKV = ((size_t)b * NKc) * C + (size_t)h * HD;

  // ---- async stage Q tile (128x64 bf16 = 1024 x b128) ----
  #pragma unroll
  for (int j = 0; j < 4; j++) {
    int f = tid + j * 256;
    int row = f >> 3, c8 = (f & 7) * 8;
    async_ld_b128(lds_addr(&Qs[row * LQ + c8]),
                  qp + baseQ + (size_t)row * C + c8);
  }
  // ---- async stage K chunk 0 (64x64 = 512 x b128) ----
  #pragma unroll
  for (int j = 0; j < 2; j++) {
    int f = tid + j * 256;
    int row = f >> 3, c8 = (f & 7) * 8;
    async_ld_b128(lds_addr(&Ks[0][row * LK + c8]),
                  kp + baseKV + (size_t)row * C + c8);
  }
  // ---- stage V chunk 0 transposed (register path) ----
  #pragma unroll
  for (int j = 0; j < 4; j++) {
    int f = tid + j * 256;
    int row = f >> 4, c4 = (f & 15) * 4;
    u32x2 dv = *(const u32x2*)(vp + baseKV + (size_t)row * C + c4);
    Vt[0][(c4 + 0) * LV + row] = (u16)(dv.x & 0xFFFFu);
    Vt[0][(c4 + 1) * LV + row] = (u16)(dv.x >> 16);
    Vt[0][(c4 + 2) * LV + row] = (u16)(dv.y & 0xFFFFu);
    Vt[0][(c4 + 3) * LV + row] = (u16)(dv.y >> 16);
  }
  wait_async0();
  __syncthreads();

  // ---- persistent Q^T B-fragments ----
  Frag qf[2];
  #pragma unroll
  for (int s = 0; s < 2; s++) {
    int qq = wid * 16 + l16;
    qf[s].q[0] = *(const u32x4*)&Qs[qq * LQ + s * 32 + half * 8];
    qf[s].q[1] = *(const u32x4*)&Qs[qq * LQ + s * 32 + half * 8 + 16];
  }

  Acc oacc[4];
  #pragma unroll
  for (int nt = 0; nt < 4; nt++)
    #pragma unroll
    for (int r = 0; r < 8; r++) oacc[nt].f[r] = 0.f;

  float mrun = -1e30f, lrun = 0.f;
  const float scale = 0.125f;   // 1/sqrt(64)

  for (int c = 0; c < NCH; c++) {
    const int p = c & 1;
    const bool more = (c + 1 < NCH);
    const int kcn = (c + 1) * KCH;

    // ---- issue next chunk's traffic: async K, register V ----
    u32x2 vR[4];
    if (more) {
      #pragma unroll
      for (int j = 0; j < 2; j++) {
        int f = tid + j * 256;
        int row = f >> 3, c8 = (f & 7) * 8;
        async_ld_b128(lds_addr(&Ks[1 - p][row * LK + c8]),
                      kp + baseKV + (size_t)(kcn + row) * C + c8);
      }
      #pragma unroll
      for (int j = 0; j < 4; j++) {
        int f = tid + j * 256;
        int row = f >> 4, c4 = (f & 15) * 4;
        vR[j] = *(const u32x2*)(vp + baseKV + (size_t)(kcn + row) * C + c4);
      }
    }

    // ---- S^T = K * Q^T : 4 m-tiles of 16 k-rows ----
    Acc sacc[4];
    #pragma unroll
    for (int mt = 0; mt < 4; mt++) {
      #pragma unroll
      for (int r = 0; r < 8; r++) sacc[mt].f[r] = 0.f;
      int row = mt * 16 + l16;
      #pragma unroll
      for (int s = 0; s < 2; s++) {
        Frag a;
        a.q[0] = *(const u32x4*)&Ks[p][row * LK + s * 32 + half * 8];
        a.q[1] = *(const u32x4*)&Ks[p][row * LK + s * 32 + half * 8 + 16];
        sacc[mt].v = __builtin_amdgcn_wmma_f32_16x16x32_bf16(
            false, a.v, false, qf[s].v, (short)0, sacc[mt].v, false, false);
      }
    }

    // ---- online softmax over column q = l16 ----
    float cmax = -1e30f;
    #pragma unroll
    for (int mt = 0; mt < 4; mt++)
      #pragma unroll
      for (int r = 0; r < 8; r++) cmax = fmaxf(cmax, sacc[mt].f[r] * scale);
    cmax = fmaxf(cmax, __shfl_xor(cmax, 16));
    float nm   = fmaxf(mrun, cmax);
    float corr = __expf(mrun - nm);
    float csum = 0.f;
    #pragma unroll
    for (int mt = 0; mt < 4; mt++)
      #pragma unroll
      for (int r = 0; r < 8; r++) {
        float e = __expf(sacc[mt].f[r] * scale - nm);
        sacc[mt].f[r] = e;
        csum += e;
      }
    csum += __shfl_xor(csum, 16);
    lrun = lrun * corr + csum;
    mrun = nm;

    // ---- rescale running O (factor indexed by q = r + 8*half via shfl) ----
    #pragma unroll
    for (int r = 0; r < 8; r++) {
      float cf = __shfl(corr, r + 8 * half);
      #pragma unroll
      for (int nt = 0; nt < 4; nt++) oacc[nt].f[r] *= cf;
    }

    // ---- P A-fragments straight from S^T accumulators ----
    Frag pf[2];
    #pragma unroll
    for (int s = 0; s < 2; s++)
      #pragma unroll
      for (int j = 0; j < 4; j++) {
        pf[s].u[j]     = packbf2(sacc[2*s    ].f[2*j], sacc[2*s    ].f[2*j+1]);
        pf[s].u[4 + j] = packbf2(sacc[2*s + 1].f[2*j], sacc[2*s + 1].f[2*j+1]);
      }

    // ---- O += P * V ----
    #pragma unroll
    for (int nt = 0; nt < 4; nt++) {
      int d = nt * 16 + l16;
      #pragma unroll
      for (int s = 0; s < 2; s++) {
        Frag bv;
        bv.q[0] = *(const u32x4*)&Vt[p][d * LV + s * 32 + half * 8];
        bv.q[1] = *(const u32x4*)&Vt[p][d * LV + s * 32 + half * 8 + 16];
        oacc[nt].v = __builtin_amdgcn_wmma_f32_16x16x32_bf16(
            false, pf[s].v, false, bv.v, (short)0, oacc[nt].v, false, false);
      }
    }

    // ---- commit next chunk's V (transpose) to buffer 1-p ----
    if (more) {
      #pragma unroll
      for (int j = 0; j < 4; j++) {
        int f = tid + j * 256;
        int row = f >> 4, c4 = (f & 15) * 4;
        Vt[1 - p][(c4 + 0) * LV + row] = (u16)(vR[j].x & 0xFFFFu);
        Vt[1 - p][(c4 + 1) * LV + row] = (u16)(vR[j].x >> 16);
        Vt[1 - p][(c4 + 2) * LV + row] = (u16)(vR[j].y & 0xFFFFu);
        Vt[1 - p][(c4 + 3) * LV + row] = (u16)(vR[j].y >> 16);
      }
      wait_async0();
    }
    __syncthreads();
  }

  // ---- finalize: divide by row sum, store bf16 (recombined-head layout) ----
  float inv = 1.f / lrun;
  #pragma unroll
  for (int r = 0; r < 8; r++) {
    float iv = __shfl(inv, r + 8 * half);
    int qrow = q0 + wid * 16 + r + 8 * half;
    #pragma unroll
    for (int nt = 0; nt < 4; nt++) {
      int d = nt * 16 + l16;
      op[((size_t)b * NQc + qrow) * C + (size_t)h * HD + d] =
          packbf1(oacc[nt].f[r] * iv);
    }
  }
}

// ---------------------------------------------------------------------------
extern "C" void kernel_launch(void* const* d_in, const int* in_sizes, int n_in,
                              void* d_out, int out_size, void* d_ws, size_t ws_size,
                              hipStream_t stream)
{
  const float* q  = (const float*)d_in[0];
  const float* k  = (const float*)d_in[1];
  const float* v  = (const float*)d_in[2];
  const float* Wq = (const float*)d_in[3];
  const float* bq = (const float*)d_in[4];
  const float* Wk = (const float*)d_in[5];
  const float* bk = (const float*)d_in[6];
  const float* Wv = (const float*)d_in[7];
  const float* bv = (const float*)d_in[8];
  const float* Wo = (const float*)d_in[9];
  const float* bo = (const float*)d_in[10];

  const int M = 2 * 2048, K = 1024, N = 1024;
  u16* qp = (u16*)d_ws;                    // [4096,1024] bf16
  u16* kp = qp + (size_t)M * N;
  u16* vp = kp + (size_t)M * N;
  u16* ow = vp + (size_t)M * N;            // attention output, bf16

  dim3 gg(N / 128, M / 128);               // (8, 32)
  gemm_bias_kernel<false, true><<<gg, 256, 0, stream>>>(q, Wq, bq, qp, M, K, N);
  gemm_bias_kernel<false, true><<<gg, 256, 0, stream>>>(k, Wk, bk, kp, M, K, N);
  gemm_bias_kernel<false, true><<<gg, 256, 0, stream>>>(v, Wv, bv, vp, M, K, N);

  attention_kernel<<<dim3(2048 / 128, 2 * 16), 256, 0, stream>>>(qp, kp, vp, ow);

  gemm_bias_kernel<true, false><<<gg, 256, 0, stream>>>(ow, Wo, bo, d_out, M, K, N);
}
